// RNN_18416819765880
// MI455X (gfx1250) — compile-verified, matrix operand order checked by reading, not run on hardware
//
#include <hip/hip_runtime.h>
#include <hip/hip_bf16.h>

// ---------------------------------------------------------------------------
// 2-layer tanh RNN on gfx1250 (MI455X), f16 WMMA with f32 accumulation.
//   x: [512,64,1024] f32, Wi/Wh: [2,1024,1024] f32, bi/bh: [2,1024] f32
//   out = concat( seq_out [512,64,1024] f32 , hs [2,64,1024] f32 )
// ---------------------------------------------------------------------------

typedef __attribute__((ext_vector_type(16))) _Float16 v16h;
typedef __attribute__((ext_vector_type(8)))  _Float16 v8h;
typedef __attribute__((ext_vector_type(4)))  _Float16 v4h;
typedef __attribute__((ext_vector_type(8)))  float    v8f;

#define T_STEPS 512
#define BATCH   64
#define HC      1024
#define ROWS    (T_STEPS * BATCH)      // 32768
#define STEP_ELEMS (BATCH * HC)        // 65536

// ---------------------------------------------------------------------------
// Fragment loader: 16 rows x 32 K of f16 from a row-major [rows][ld] matrix.
// CDNA5 16-bit A-matrix layout (ISA 7.12.2): lane L in [0,16) holds row L,
// K = {0..7, 16..23}; lane L+16 holds row L, K = {8..15, 24..31}.
// The B matrix (stored [N][K] row-major, i.e. W[h_out][k]) uses the identical
// pattern with lane -> output column.
// ---------------------------------------------------------------------------
__device__ __forceinline__ v16h load_frag(const _Float16* __restrict__ base,
                                          int row, int k, int ld) {
  const int lane = threadIdx.x & 31;
  const int r    = row + (lane & 15);
  const int koff = k + ((lane & 16) ? 8 : 0);
  const _Float16* q = base + (size_t)r * ld + koff;
  v8h lo = *(const v8h*)(q);        // K = koff .. koff+7
  v8h hi = *(const v8h*)(q + 16);   // K = koff+16 .. koff+23
  return __builtin_shufflevector(lo, hi, 0,1,2,3,4,5,6,7,8,9,10,11,12,13,14,15);
}

// Same fragment pattern but from an LDS-resident [64][1024] f16 tile.
__device__ __forceinline__ v16h load_frag_lds(const _Float16* base,
                                              int row, int k) {
  const int lane = threadIdx.x & 31;
  const _Float16* q = base + (row + (lane & 15)) * HC + k + ((lane & 16) ? 8 : 0);
  v8h lo = *(const v8h*)(q);
  v8h hi = *(const v8h*)(q + 16);
  return __builtin_shufflevector(lo, hi, 0,1,2,3,4,5,6,7,8,9,10,11,12,13,14,15);
}

__device__ __forceinline__ v8f wmma_f16(v16h a, v16h b, v8f c) {
  return __builtin_amdgcn_wmma_f32_16x16x32_f16(false, a, false, b,
                                                (short)0, c, false, false);
}

// ---------------------------------------------------------------------------
// f32 -> f16 conversion, 4 elements/thread
// ---------------------------------------------------------------------------
__global__ void cvt_f32_to_f16(const float* __restrict__ in,
                               _Float16* __restrict__ out, int n4) {
  int i = blockIdx.x * blockDim.x + threadIdx.x;
  if (i < n4) {
    float4 v = ((const float4*)in)[i];
    v4h o;
    o[0] = (_Float16)v.x; o[1] = (_Float16)v.y;
    o[2] = (_Float16)v.z; o[3] = (_Float16)v.w;
    ((v4h*)out)[i] = o;
  }
}

// ---------------------------------------------------------------------------
// pre = A @ W^T + bias.   A: [32768][1024] f16 row-major.
// W: [1024][1024] f16 row-major ([N][K]).  pre: [32768][1024] f32.
// Block = 256 thr (8 waves), tile 64(M) x 256(N); wave tile 32x64.
// Grid: (ROWS/64, HC/256) = (512, 4).
// ---------------------------------------------------------------------------
__global__ void gemm_pre(const _Float16* __restrict__ A,
                         const _Float16* __restrict__ W,
                         const float* __restrict__ bias,
                         float* __restrict__ P) {
  const int lane = threadIdx.x & 31;
  const int w    = threadIdx.x >> 5;
  const int wm   = w & 1;        // 2 M-waves
  const int wn   = w >> 1;       // 4 N-waves
  const int rowBase = blockIdx.x * 64  + wm * 32;
  const int colBase = blockIdx.y * 256 + wn * 64;

  v8f c[2][4];
#pragma unroll
  for (int i = 0; i < 2; ++i)
#pragma unroll
    for (int j = 0; j < 4; ++j) c[i][j] = (v8f)(0.0f);

#pragma unroll 2
  for (int k0 = 0; k0 < HC; k0 += 32) {
    v16h a0 = load_frag(A, rowBase,      k0, HC);
    v16h a1 = load_frag(A, rowBase + 16, k0, HC);
    v16h b0 = load_frag(W, colBase +  0, k0, HC);
    v16h b1 = load_frag(W, colBase + 16, k0, HC);
    v16h b2 = load_frag(W, colBase + 32, k0, HC);
    v16h b3 = load_frag(W, colBase + 48, k0, HC);
    c[0][0] = wmma_f16(a0, b0, c[0][0]);
    c[0][1] = wmma_f16(a0, b1, c[0][1]);
    c[0][2] = wmma_f16(a0, b2, c[0][2]);
    c[0][3] = wmma_f16(a0, b3, c[0][3]);
    c[1][0] = wmma_f16(a1, b0, c[1][0]);
    c[1][1] = wmma_f16(a1, b1, c[1][1]);
    c[1][2] = wmma_f16(a1, b2, c[1][2]);
    c[1][3] = wmma_f16(a1, b3, c[1][3]);
  }

  // C/D layout: col = lane&15; row = vgpr index e + (lane>=16 ? 8 : 0)
  const int rOff = (lane & 16) ? 8 : 0;
  const int cOff = lane & 15;
#pragma unroll
  for (int i = 0; i < 2; ++i) {
#pragma unroll
    for (int j = 0; j < 4; ++j) {
#pragma unroll
      for (int e = 0; e < 8; ++e) {
        int r = rowBase + i * 16 + rOff + e;
        int cc = colBase + j * 16 + cOff;
        P[(size_t)r * HC + cc] = c[i][j][e] + bias[cc];
      }
    }
  }
}

// ---------------------------------------------------------------------------
// Persistent recurrent scan for one layer.
//   for t in 0..511:  h[t+1] = tanh(pre[t] + h[t] @ Wh^T + bh)
// hseq: [(T+1)*64][1024] f16; row block 0 is zeros (h0). Step t reads block t,
// writes block t+1. Grid-wide barrier (monotonic counter) between steps.
// Grid: 16 blocks x 256 thr. Block owns 64 output cols; its Wh slice
// (64 x 1024 f16 = 128 KB) is staged once into LDS (320 KB/WGP available) so
// every sequential step reads weights at LDS latency instead of L2.
// Wave w handles M-tile (w&3) and two N-tiles (w>>2)*2 .. +1, full K=1024.
// ---------------------------------------------------------------------------
__global__ void rnn_scan(const float* __restrict__ pre,
                         const _Float16* __restrict__ Wh,
                         const float* __restrict__ bh,
                         _Float16* __restrict__ hseq,
                         float* __restrict__ outSeq,   // may be null (layer 0)
                         float* __restrict__ hsOut,
                         unsigned* __restrict__ counter) {
  extern __shared__ _Float16 ldsW[];   // [64][1024] f16 = 128 KB (dynamic)

  const int lane = threadIdx.x & 31;
  const int w    = threadIdx.x >> 5;
  const int mi   = w & 3;               // M-tile: rows 16*mi .. +15  (batch)
  const int nj   = (w >> 2) * 2;        // first of two N-tiles
  const int colBase = blockIdx.x * 64;
  const int rOff = (lane & 16) ? 8 : 0;
  const int cOff = lane & 15;

  // ---- stage this block's Wh slice (rows colBase..+63, contiguous) in LDS --
  {
    const v8h* src = (const v8h*)(Wh + (size_t)colBase * HC);
    v8h* dst = (v8h*)ldsW;
    const int nvec = (64 * HC) / 8;     // 8192 x 16B
    for (int i = threadIdx.x; i < nvec; i += blockDim.x) dst[i] = src[i];
  }
  __syncthreads();

  // t-invariant values off the critical path
  const int cL0 = nj * 16 + cOff;       // column within LDS slice
  const int cL1 = cL0 + 16;
  const int c0 = colBase + cL0;         // global column
  const int c1 = colBase + cL1;
  const float bh0 = bh[c0];
  const float bh1 = bh[c1];

  unsigned target = 0;
  for (int t = 0; t < T_STEPS; ++t) {
    const _Float16* h = hseq + (size_t)t * STEP_ELEMS;
    const float* preT = pre + (size_t)t * STEP_ELEMS;

    // issue pre[t] loads early so they overlap the WMMA K-loop
    float pv0[8], pv1[8];
#pragma unroll
    for (int e = 0; e < 8; ++e) {
      int b = mi * 16 + rOff + e;
      pv0[e] = preT[b * HC + c0];
      pv1[e] = preT[b * HC + c1];
    }

    v8f acc0 = (v8f)(0.0f);
    v8f acc1 = (v8f)(0.0f);
#pragma unroll 4
    for (int k0 = 0; k0 < HC; k0 += 32) {
      v16h a  = load_frag(h, mi * 16, k0, HC);       // global (h changes / step)
      v16h b0 = load_frag_lds(ldsW, nj * 16,      k0); // LDS-resident weights
      v16h b1 = load_frag_lds(ldsW, nj * 16 + 16, k0);
      acc0 = wmma_f16(a, b0, acc0);
      acc1 = wmma_f16(a, b1, acc1);
    }

    _Float16* hNext = hseq + (size_t)(t + 1) * STEP_ELEMS;
#pragma unroll
    for (int e = 0; e < 8; ++e) {
      int b = mi * 16 + rOff + e;       // batch row 0..63
      {
        float v = tanhf(pv0[e] + acc0[e] + bh0);
        hNext[b * HC + c0] = (_Float16)v;
        if (outSeq) outSeq[(size_t)t * STEP_ELEMS + b * HC + c0] = v;
        if (t == T_STEPS - 1) hsOut[b * HC + c0] = v;
      }
      {
        float v = tanhf(pv1[e] + acc1[e] + bh1);
        hNext[b * HC + c1] = (_Float16)v;
        if (outSeq) outSeq[(size_t)t * STEP_ELEMS + b * HC + c1] = v;
        if (t == T_STEPS - 1) hsOut[b * HC + c1] = v;
      }
    }

    // ---- device-wide barrier (monotonic counter, no reset races) ----
    target += gridDim.x;
    __threadfence();
    __syncthreads();
    if (threadIdx.x == 0) {
      atomicAdd(counter, 1u);
      while (*(volatile unsigned*)counter < target) {
        __builtin_amdgcn_s_sleep(1);
      }
    }
    __syncthreads();
    __threadfence();
  }
}

// ---------------------------------------------------------------------------
// Workspace layout (bytes):
//   WiH   f16 [2*1024*1024]          @ 0          (4,194,304)
//   WhH   f16 [2*1024*1024]          @ 4,194,304  (4,194,304)
//   xH    f16 [32768*1024]           @ 8,388,608  (67,108,864)
//   hseq  f16 [(512+1)*64*1024]      @ 75,497,472 (67,239,936)
//   pre   f32 [32768*1024]           @ 142,737,408 (134,217,728)
//   ctr   u32 [2]                    @ 276,955,136
// ---------------------------------------------------------------------------
extern "C" void kernel_launch(void* const* d_in, const int* in_sizes, int n_in,
                              void* d_out, int out_size, void* d_ws, size_t ws_size,
                              hipStream_t stream) {
  const float* x  = (const float*)d_in[0];
  const float* Wi = (const float*)d_in[1];
  const float* bi = (const float*)d_in[2];
  const float* Wh = (const float*)d_in[3];
  const float* bh = (const float*)d_in[4];

  char* ws = (char*)d_ws;
  _Float16* WiH  = (_Float16*)(ws + 0);
  _Float16* WhH  = (_Float16*)(ws + 4194304);
  _Float16* xH   = (_Float16*)(ws + 8388608);
  _Float16* hseq = (_Float16*)(ws + 75497472);
  float*    pre  = (float*)   (ws + 142737408);
  unsigned* ctr  = (unsigned*)(ws + 276955136);

  float* outSeq = (float*)d_out;                       // [512,64,1024]
  float* hsOut  = (float*)d_out + (size_t)ROWS * HC;   // [2,64,1024]

  // Zero barrier counters and h0 row (deterministic each call).
  hipMemsetAsync(ctr, 0, 2 * sizeof(unsigned), stream);
  hipMemsetAsync(hseq, 0, (size_t)STEP_ELEMS * sizeof(_Float16), stream);

  // f32 -> f16 conversions.
  {
    int n4 = (2 * HC * HC) / 4;   // 524288
    cvt_f32_to_f16<<<(n4 + 255) / 256, 256, 0, stream>>>(Wi, WiH, n4);
    cvt_f32_to_f16<<<(n4 + 255) / 256, 256, 0, stream>>>(Wh, WhH, n4);
    int m4 = (ROWS * HC) / 4;     // 8388608
    cvt_f32_to_f16<<<(m4 + 255) / 256, 256, 0, stream>>>(x, xH, m4);
  }

  dim3 ggrid(ROWS / 64, HC / 256);   // (512, 4)
  const size_t scanLds = 64 * HC * sizeof(_Float16);   // 128 KB dynamic LDS

  // ---- Layer 0 ----
  gemm_pre<<<ggrid, 256, 0, stream>>>(xH, WiH, bi, pre);
  rnn_scan<<<16, 256, scanLds, stream>>>(pre, WhH, bh, hseq,
                                         /*outSeq=*/nullptr, hsOut, ctr + 0);

  // ---- Layer 1 ----  (A-matrix = layer-0 hidden sequence, already f16)
  gemm_pre<<<ggrid, 256, 0, stream>>>(hseq + STEP_ELEMS, WiH + HC * HC,
                                      bi + HC, pre);
  rnn_scan<<<16, 256, scanLds, stream>>>(pre, WhH + HC * HC, bh + HC, hseq,
                                         outSeq, hsOut + STEP_ELEMS, ctr + 1);
}
